// Transducer_48765058678800
// MI455X (gfx1250) — compile-verified, matrix operand order checked by reading, not run on hardware
//
#include <hip/hip_runtime.h>

// ---------------------------------------------------------------------------
// RNN-Transducer forward loss on MI455X (gfx1250), wave32 + WMMA bf16.
//
//   1. pack_b_kernel      : f32 weights -> bf16 WMMA B-operand tiles
//   2. lstm_kernel  x7    : persistent-WG LSTM (audio x6, label x1), WMMA gates
//   3. proj_kernel  x2    : enc/dec linear projections (WMMA)
//   4. joint_kernel       : fused joint GEMM + tied-emb GEMM + online log-softmax
//   5. rnnt_loss_kernel   : forward-variable DP -> scalar loss
//
// Hot loops use: hoisted per-lane pointers (constant stride per K-step),
// dual accumulators (2 N-tiles / iter) to fill WMMA hazard slots, branchless
// A-operand ds_load_b128 pairs, and global_prefetch for the next tile pair.
// ---------------------------------------------------------------------------

typedef __bf16 bf16_t;
typedef __attribute__((ext_vector_type(16))) __bf16 v16bf;
typedef __attribute__((ext_vector_type(8)))  float  v8f;

#define Bc   4
#define Tt   200
#define Uu   30
#define U1   31
#define INd  80
#define EHd  512
#define EOd  340
#define DHd  512
#define DOd  320
#define JDd  512
#define Vv   4232
#define NGATE 2048
#define NTV2 266          // ceil(4232/16)=265, padded to even

__device__ __forceinline__ bf16_t f2bf(float f) { return (bf16_t)f; }

struct alignas(16) H8 { bf16_t h[8]; };

// B-operand tile: 16 contiguous halves at p (lane pre-folded into p).
__device__ __forceinline__ v16bf load_bp(const bf16_t* p) {
  H8 lo = *reinterpret_cast<const H8*>(p);
  H8 hi = *reinterpret_cast<const H8*>(p + 8);
  v16bf r;
#pragma unroll
  for (int e = 0; e < 8; ++e) { r[e] = lo.h[e]; r[8 + e] = hi.h[e]; }
  return r;
}

// A-operand from LDS: halves [0..7] at p, [8..15] at p+16 (ISA 16-bit A layout:
// lane l -> m=l&15, khalf=(l>>4)*8, element e -> K=(e>>3)*16+khalf+(e&7)).
__device__ __forceinline__ v16bf load_ap(const bf16_t* p) {
  H8 lo = *reinterpret_cast<const H8*>(p);
  H8 hi = *reinterpret_cast<const H8*>(p + 16);
  v16bf r;
#pragma unroll
  for (int e = 0; e < 8; ++e) { r[e] = lo.h[e]; r[8 + e] = hi.h[e]; }
  return r;
}

#define WMMA_BF16(a, b, c) \
  __builtin_amdgcn_wmma_f32_16x16x32_bf16(false, (a), false, (b), (short)0, (c), false, false)

// ---------------------------------------------------------------------------
// Pack B operand: logical B[k][n] = concat(W1,W2) row n, col k (W^T), zero pad.
// tileIdx = kt*Ntiles + nt; each tile = 32 lanes x 16 halves contiguous.
// ---------------------------------------------------------------------------
__global__ void pack_b_kernel(const float* __restrict__ W1, int K1,
                              const float* __restrict__ W2, int K2,
                              bf16_t* __restrict__ out, int N, int Ktiles, int Ntiles) {
  int idx = blockIdx.x * blockDim.x + threadIdx.x;
  int total = Ktiles * Ntiles * 32;
  if (idx >= total) return;
  int lane = idx & 31;
  int tile = idx >> 5;
  int nt = tile % Ntiles;
  int kt = tile / Ntiles;
  int n = nt * 16 + (lane & 15);
  int khalf = (lane >> 4) * 8;
  bf16_t* o = out + (long)idx * 16;
#pragma unroll
  for (int e = 0; e < 16; ++e) {
    int k = kt * 32 + ((e >> 3) << 4) + khalf + (e & 7);
    float v = 0.f;
    if (n < N) {
      if (k < K1)            v = W1[(long)n * K1 + k];
      else if (k - K1 < K2)  v = W2[(long)n * K2 + (k - K1)];
    }
    o[e] = f2bf(v);
  }
}

// ---------------------------------------------------------------------------
__global__ void embed_kernel(const int* __restrict__ targets, const float* __restrict__ emb,
                             float* __restrict__ elab) {
  int i = blockIdx.x * blockDim.x + threadIdx.x;
  if (i >= Bc * U1 * DHd) return;
  int k = i & 511;
  int bu = i >> 9;
  int u = bu % U1;
  int b = bu / U1;
  int tok = (u == 0) ? 0 : targets[b * Uu + (u - 1)];
  elab[i] = emb[(long)tok * DHd + k];
}

// ---------------------------------------------------------------------------
// Persistent-workgroup LSTM, 8 waves. A panel has a 5th all-zero row and the
// loader clamps m -> branchless ds loads for the 16x? WMMA A operand.
// ---------------------------------------------------------------------------
#define LSTM_THREADS 256
__global__ void lstm_kernel(const float* __restrict__ xin, int inD, int Tlen,
                            const bf16_t* __restrict__ wcat, int Ktiles,
                            const float* __restrict__ bias,
                            float* __restrict__ hout) {
  __shared__ bf16_t A[5 * 1024];           // rows 0..3 live, row 4 stays zero
  __shared__ float h[4 * EHd], c[4 * EHd];
  __shared__ float G[4 * NGATE];
  int tid = threadIdx.x;
  int wave = tid >> 5, lane = tid & 31;
  int Kp = Ktiles * 32;

  for (int i = tid; i < 5 * 1024; i += LSTM_THREADS) A[i] = f2bf(0.f);
  for (int i = tid; i < 4 * EHd; i += LSTM_THREADS) { h[i] = 0.f; c[i] = 0.f; }
  __syncthreads();

  int m = lane & 15;
  int mm = m < 4 ? m : 4;                       // clamp to zero row
  const bf16_t* ap = A + mm * Kp + ((lane >> 4) * 8);
  int hi = lane >> 4;

  for (int t = 0; t < Tlen; ++t) {
    for (int i = tid; i < 4 * Kp; i += LSTM_THREADS) {
      int r = i / Kp, k = i - r * Kp;
      float v = 0.f;
      if (k < inD)             v = xin[((long)r * Tlen + t) * inD + k];
      else if (k - inD < EHd)  v = h[r * EHd + (k - inD)];
      A[r * Kp + k] = f2bf(v);
    }
    __syncthreads();
    // 64 N-tile pairs over 8 waves; dual accumulators share the A operand.
    for (int pr = wave; pr < 64; pr += 8) {
      int nt0 = pr * 2;
      const bf16_t* bp = wcat + ((long)nt0 * 32 + lane) * 16;
      v8f acc0 = {}, acc1 = {};
#pragma unroll 4
      for (int kt = 0; kt < Ktiles; ++kt) {
        v16bf a  = load_ap(ap + kt * 32);
        v16bf b0 = load_bp(bp);
        v16bf b1 = load_bp(bp + 512);
        acc0 = WMMA_BF16(a, b0, acc0);
        acc1 = WMMA_BF16(a, b1, acc1);
        bp += (long)128 * 512;
      }
      int n0 = nt0 * 16 + m;
#pragma unroll
      for (int r = 0; r < 8; ++r) {
        int mr = r + 8 * hi;
        if (mr < 4) {
          G[mr * NGATE + n0]      = acc0[r];
          G[mr * NGATE + n0 + 16] = acc1[r];
        }
      }
    }
    __syncthreads();
    for (int i = tid; i < 4 * EHd; i += LSTM_THREADS) {
      int b_ = i >> 9, j = i & 511;
      float gi = G[b_ * NGATE + j]        + bias[j];
      float gf = G[b_ * NGATE + 512 + j]  + bias[512 + j];
      float gg = G[b_ * NGATE + 1024 + j] + bias[1024 + j];
      float go = G[b_ * NGATE + 1536 + j] + bias[1536 + j];
      float si = 1.f / (1.f + __expf(-gi));
      float sf = 1.f / (1.f + __expf(-gf));
      float so = 1.f / (1.f + __expf(-go));
      float cc = sf * c[i] + si * tanhf(gg);
      float hh = so * tanhf(cc);
      c[i] = cc; h[i] = hh;
      hout[((long)b_ * Tlen + t) * EHd + j] = hh;
    }
    __syncthreads();
  }
}

// ---------------------------------------------------------------------------
// Linear projection: out[M,N] = src[M,512] @ W^T + b. One wave per 16-row tile.
// Ntiles must be even (22 / 20 here).
// ---------------------------------------------------------------------------
__global__ void proj_kernel(const float* __restrict__ src, int M,
                            const bf16_t* __restrict__ wpk, int Ntiles,
                            const float* __restrict__ bias, int N,
                            float* __restrict__ out) {
  __shared__ bf16_t A[16 * 512];
  int lane = threadIdx.x & 31;
  int m0 = blockIdx.x * 16;
  for (int i = threadIdx.x; i < 16 * 512; i += 32) {
    int r = i >> 9, k = i & 511;
    int mg = m0 + r;
    A[i] = f2bf(mg < M ? src[(long)mg * 512 + k] : 0.f);
  }
  __syncthreads();
  const bf16_t* ap = A + (lane & 15) * 512 + ((lane >> 4) * 8);
  int hi = lane >> 4, nl = lane & 15;
  for (int pr = 0; pr < Ntiles / 2; ++pr) {
    int nt0 = pr * 2;
    const bf16_t* bp = wpk + ((long)nt0 * 32 + lane) * 16;
    v8f acc0 = {}, acc1 = {};
#pragma unroll 4
    for (int kt = 0; kt < 16; ++kt) {
      v16bf a  = load_ap(ap + kt * 32);
      v16bf b0 = load_bp(bp);
      v16bf b1 = load_bp(bp + 512);
      acc0 = WMMA_BF16(a, b0, acc0);
      acc1 = WMMA_BF16(a, b1, acc1);
      bp += (long)Ntiles * 512;
    }
    int n0 = nt0 * 16 + nl;
#pragma unroll
    for (int r = 0; r < 8; ++r) {
      int mg = m0 + r + 8 * hi;
      if (mg < M) {
        if (n0 < N)      out[(long)mg * N + n0]      = acc0[r] + bias[n0];
        if (n0 + 16 < N) out[(long)mg * N + n0 + 16] = acc1[r] + bias[n0 + 16];
      }
    }
  }
}

// ---------------------------------------------------------------------------
// Fused joint network + tied-weight output + online log-softmax.
// One block (4 waves) per 16 lattice points. flat = (b*T+t)*31 + u.
// ---------------------------------------------------------------------------
#define JBLK 128
#define KCAT 672          // 660 padded to 21 K-tiles
__global__ void joint_kernel(const float* __restrict__ enc, const float* __restrict__ dec,
                             const bf16_t* __restrict__ wjoint, const float* __restrict__ jointb,
                             const bf16_t* __restrict__ wembB, const float* __restrict__ outb,
                             const int* __restrict__ targets,
                             float* __restrict__ lpb, float* __restrict__ lpl) {
  __shared__ bf16_t CAT[16 * KCAT];
  __shared__ bf16_t JT[16 * JDd];
  __shared__ float stash_blank[16];
  __shared__ float stash_lab[16];
  __shared__ float wmax[4 * 16], wsum[4 * 16];

  int tid = threadIdx.x, wave = tid >> 5, lane = tid & 31;
  long tile0 = (long)blockIdx.x * 16;   // 24800 = 1550*16 exactly

  for (int i = tid; i < 16 * KCAT; i += JBLK) {
    int r = i / KCAT, k = i - r * KCAT;
    long flat = tile0 + r;
    int u = (int)(flat % U1);
    long bt = flat / U1;
    float v = 0.f;
    if (k < EOd)              v = enc[bt * EOd + k];
    else if (k < EOd + DOd)   v = dec[((bt / Tt) * U1 + u) * DOd + (k - EOd)];
    CAT[i] = f2bf(v);
  }
  if (tid < 16) { stash_blank[tid] = 0.f; stash_lab[tid] = 0.f; }
  __syncthreads();

  int hi = lane >> 4, nl = lane & 15;

  // ---- Stage 1: JT = tanh([enc|dec] @ joint_W^T + b), 16 N-tile pairs / 4 waves
  {
    const bf16_t* ap = CAT + nl * KCAT + hi * 8;
    for (int pr = wave; pr < 16; pr += 4) {
      int nt0 = pr * 2;
      const bf16_t* bp = wjoint + ((long)nt0 * 32 + lane) * 16;
      v8f acc0 = {}, acc1 = {};
#pragma unroll 3
      for (int kt = 0; kt < KCAT / 32; ++kt) {
        v16bf a  = load_ap(ap + kt * 32);
        v16bf b0 = load_bp(bp);
        v16bf b1 = load_bp(bp + 512);
        acc0 = WMMA_BF16(a, b0, acc0);
        acc1 = WMMA_BF16(a, b1, acc1);
        bp += (long)32 * 512;
      }
      int n0 = nt0 * 16 + nl;
#pragma unroll
      for (int r = 0; r < 8; ++r) {
        int mr = r + 8 * hi;
        JT[mr * JDd + n0]      = f2bf(tanhf(acc0[r] + jointb[n0]));
        JT[mr * JDd + n0 + 16] = f2bf(tanhf(acc1[r] + jointb[n0 + 16]));
      }
    }
  }
  __syncthreads();

  // ---- Stage 2: stream 133 V-tile pairs with online log-softmax
  // Per-row loop invariants (target token per lattice row).
  int tokrow[8];
#pragma unroll
  for (int r = 0; r < 8; ++r) {
    long flat = tile0 + (r + 8 * hi);
    int u = (int)(flat % U1);
    long bt = flat / U1;
    tokrow[r] = (u < Uu) ? targets[(bt / Tt) * Uu + u] : -1;
  }

  float rmax[8], rsum[8];
#pragma unroll
  for (int r = 0; r < 8; ++r) { rmax[r] = -INFINITY; rsum[r] = 0.f; }

  const bf16_t* ap = JT + nl * JDd + hi * 8;
  const long kstr = (long)NTV2 * 512;

  for (int pr = wave; pr < NTV2 / 2; pr += 4) {
    int nt0 = pr * 2;
    const bf16_t* bp = wembB + ((long)nt0 * 32 + lane) * 16;
    if (pr + 4 < NTV2 / 2)
      __builtin_prefetch(wembB + ((long)(nt0 + 8) * 32 + lane) * 16, 0, 3);
    v8f acc0 = {}, acc1 = {};
#pragma unroll 4
    for (int kt = 0; kt < 16; ++kt) {
      v16bf a  = load_ap(ap + kt * 32);
      v16bf b0 = load_bp(bp);
      v16bf b1 = load_bp(bp + 512);
      acc0 = WMMA_BF16(a, b0, acc0);
      acc1 = WMMA_BF16(a, b1, acc1);
      bp += kstr;
    }
    int n0 = nt0 * 16 + nl;
    int n1 = n0 + 16;
    bool v0 = n0 < Vv, v1 = n1 < Vv;
    float bb0 = v0 ? outb[n0] : 0.f;
    float bb1 = v1 ? outb[n1] : 0.f;
#pragma unroll
    for (int r = 0; r < 8; ++r) {
      int mr = r + 8 * hi;
      float x0 = v0 ? acc0[r] + bb0 : -INFINITY;
      float x1 = v1 ? acc1[r] + bb1 : -INFINITY;
      if (n0 == 0)         stash_blank[mr] = x0;
      if (n0 == tokrow[r]) stash_lab[mr]   = x0;
      if (n1 == tokrow[r]) stash_lab[mr]   = x1;
      float mo = rmax[r];
      float mn = fmaxf(mo, fmaxf(x0, x1));
      if (mn > -INFINITY) {
        rsum[r] = rsum[r] * __expf(mo - mn) + __expf(x0 - mn) + __expf(x1 - mn);
        rmax[r] = mn;
      }
    }
  }
  // reduce across the 16-lane half (columns of this row group)
  for (int off = 1; off < 16; off <<= 1) {
#pragma unroll
    for (int r = 0; r < 8; ++r) {
      float om = __shfl_xor(rmax[r], off, 32);
      float os = __shfl_xor(rsum[r], off, 32);
      float mn = fmaxf(rmax[r], om);
      rsum[r] = rsum[r] * __expf(rmax[r] - mn) + os * __expf(om - mn);
      rmax[r] = mn;
    }
  }
  if (nl == 0) {
#pragma unroll
    for (int r = 0; r < 8; ++r) {
      wmax[wave * 16 + hi * 8 + r] = rmax[r];
      wsum[wave * 16 + hi * 8 + r] = rsum[r];
    }
  }
  __syncthreads();
  if (tid < 16) {
    int mr = tid;
    float M_ = -INFINITY, S = 0.f;
    for (int w = 0; w < 4; ++w) {
      float om = wmax[w * 16 + mr], os = wsum[w * 16 + mr];
      float mn = fmaxf(M_, om);
      S = S * __expf(M_ - mn) + os * __expf(om - mn);
      M_ = mn;
    }
    float logZ = M_ + __logf(S);
    long flat = tile0 + mr;
    int u = (int)(flat % U1);
    long bt = flat / U1;
    lpb[bt * U1 + u] = stash_blank[mr] - logZ;
    if (u < Uu) lpl[bt * Uu + u] = stash_lab[mr] - logZ;
  }
}

// ---------------------------------------------------------------------------
// RNN-T forward DP: alpha recursion over (T, U+1), one thread per batch item.
// ---------------------------------------------------------------------------
__global__ void rnnt_loss_kernel(const float* __restrict__ lpb, const float* __restrict__ lpl,
                                 const int* __restrict__ tlen, const int* __restrict__ ulen,
                                 float* __restrict__ out) {
  __shared__ float ll[Bc];
  int b = threadIdx.x;
  if (b < Bc) {
    const float* LB = lpb + (long)b * Tt * U1;
    const float* LL = lpl + (long)b * Tt * Uu;
    float alpha[U1];
    alpha[0] = 0.f;
    for (int u = 1; u < U1; ++u) alpha[u] = alpha[u - 1] + LL[u - 1];
    int tl = tlen[b], ul = ulen[b];
    float saved = (tl == 1) ? alpha[ul] : 0.f;
    for (int t = 1; t < Tt; ++t) {
      float prev = alpha[0] + LB[(t - 1) * U1 + 0];
      alpha[0] = prev;
      for (int u = 1; u < U1; ++u) {
        float basu = alpha[u] + LB[(t - 1) * U1 + u];
        float cand = prev + LL[t * Uu + (u - 1)];
        float mx = fmaxf(basu, cand);
        prev = mx + __logf(__expf(basu - mx) + __expf(cand - mx));
        alpha[u] = prev;
      }
      if (t == tl - 1) saved = alpha[ul];
    }
    ll[b] = saved + LB[(tl - 1) * U1 + ul];
  }
  __syncthreads();
  if (threadIdx.x == 0) out[0] = -(ll[0] + ll[1] + ll[2] + ll[3]) * 0.25f;
}

// ---------------------------------------------------------------------------
// Host orchestration
// ---------------------------------------------------------------------------
extern "C" void kernel_launch(void* const* d_in, const int* in_sizes, int n_in,
                              void* d_out, int out_size, void* d_ws, size_t ws_size,
                              hipStream_t stream) {
  (void)in_sizes; (void)n_in; (void)out_size; (void)ws_size;
  const float* inputs = (const float*)d_in[0];
  const int*   in_len = (const int*)  d_in[1];
  const int*   targets= (const int*)  d_in[2];
  const int*   tgt_len= (const int*)  d_in[3];
  const float* aW0    = (const float*)d_in[4];
  const float* aU0    = (const float*)d_in[5];
  const float* ab0    = (const float*)d_in[6];
  const float* aW     = (const float*)d_in[7];
  const float* aU     = (const float*)d_in[8];
  const float* ab     = (const float*)d_in[9];
  const float* apW    = (const float*)d_in[10];
  const float* apb    = (const float*)d_in[11];
  const float* emb    = (const float*)d_in[12];
  const float* lW     = (const float*)d_in[13];
  const float* lU     = (const float*)d_in[14];
  const float* lb     = (const float*)d_in[15];
  const float* lpW    = (const float*)d_in[16];
  const float* lpb_   = (const float*)d_in[17];
  const float* jW     = (const float*)d_in[18];
  const float* jb     = (const float*)d_in[19];
  const float* job    = (const float*)d_in[20];

  char* p = (char*)d_ws;
  auto carve = [&](size_t bytes) -> void* {
    void* r = (void*)p;
    p += (bytes + 255) & ~(size_t)255;
    return r;
  };
  bf16_t* P0  = (bf16_t*)carve((size_t)19 * 128 * 1024);
  bf16_t* P15 = (bf16_t*)carve((size_t)5 * 32 * 128 * 1024);
  bf16_t* PL  = (bf16_t*)carve((size_t)32 * 128 * 1024);
  bf16_t* PEP = (bf16_t*)carve((size_t)16 * 22 * 1024);
  bf16_t* PDP = (bf16_t*)carve((size_t)16 * 20 * 1024);
  bf16_t* PJ  = (bf16_t*)carve((size_t)21 * 32 * 1024);
  bf16_t* PE  = (bf16_t*)carve((size_t)16 * NTV2 * 1024);
  float* hA   = (float*)carve((size_t)Bc * Tt * EHd * 4);
  float* hB   = (float*)carve((size_t)Bc * Tt * EHd * 4);
  float* elab = (float*)carve((size_t)Bc * U1 * DHd * 4);
  float* hL   = (float*)carve((size_t)Bc * U1 * DHd * 4);
  float* enc  = (float*)carve((size_t)Bc * Tt * EOd * 4);
  float* dec  = (float*)carve((size_t)Bc * U1 * DOd * 4);
  float* lpB  = (float*)carve((size_t)Bc * Tt * U1 * 4);
  float* lpL  = (float*)carve((size_t)Bc * Tt * Uu * 4);

  auto packg = [&](int kt, int ntl) { return dim3((kt * ntl * 32 + 255) / 256); };

  pack_b_kernel<<<packg(19, 128), 256, 0, stream>>>(aW0, INd, aU0, EHd, P0, NGATE, 19, 128);
  for (int l = 0; l < 5; ++l)
    pack_b_kernel<<<packg(32, 128), 256, 0, stream>>>(
        aW + (long)l * NGATE * EHd, EHd, aU + (long)l * NGATE * EHd, EHd,
        P15 + (long)l * 32 * 128 * 512, NGATE, 32, 128);
  pack_b_kernel<<<packg(32, 128), 256, 0, stream>>>(lW, DHd, lU, DHd, PL, NGATE, 32, 128);
  pack_b_kernel<<<packg(16, 22), 256, 0, stream>>>(apW, EHd, nullptr, 0, PEP, EOd, 16, 22);
  pack_b_kernel<<<packg(16, 20), 256, 0, stream>>>(lpW, DHd, nullptr, 0, PDP, DOd, 16, 20);
  pack_b_kernel<<<packg(21, 32), 256, 0, stream>>>(jW, EOd + DOd, nullptr, 0, PJ, JDd, 21, 32);
  pack_b_kernel<<<packg(16, NTV2), 256, 0, stream>>>(emb, DHd, nullptr, 0, PE, Vv, 16, NTV2);

  lstm_kernel<<<1, LSTM_THREADS, 0, stream>>>(inputs, INd, Tt, P0, 19, ab0, hA);
  const float* src = hA; float* dst = hB;
  for (int l = 0; l < 5; ++l) {
    lstm_kernel<<<1, LSTM_THREADS, 0, stream>>>(
        src, EHd, Tt, P15 + (long)l * 32 * 128 * 512, 32, ab + (long)l * NGATE, dst);
    const float* t_ = dst; dst = (float*)src; src = t_;
  }
  embed_kernel<<<(Bc * U1 * DHd + 255) / 256, 256, 0, stream>>>(targets, emb, elab);
  lstm_kernel<<<1, LSTM_THREADS, 0, stream>>>(elab, DHd, U1, PL, 32, lb, hL);

  proj_kernel<<<(Bc * Tt + 15) / 16, 32, 0, stream>>>(src, Bc * Tt, PEP, 22, apb, EOd, enc);
  proj_kernel<<<(Bc * U1 + 15) / 16, 32, 0, stream>>>(hL, Bc * U1, PDP, 20, lpb_, DOd, dec);

  joint_kernel<<<(Bc * Tt * U1) / 16, JBLK, 0, stream>>>(
      enc, dec, PJ, jb, PE, job, targets, lpB, lpL);

  rnnt_loss_kernel<<<1, 32, 0, stream>>>(lpB, lpL, in_len, tgt_len, (float*)d_out);
}
